// LSTMClassifier_2534030705278
// MI455X (gfx1250) — matrix operand order repaired and checked by reading, not compiled, not run on hardware
//
#include <hip/hip_runtime.h>
#include <hip/hip_bf16.h>
#include <stdint.h>
#include <stddef.h>

// ---------------------------------------------------------------------------
// LSTMClassifier for MI455X (gfx1250, wave32, WMMA bf16 16x16x32).
//   H=64, IN=4, B=256, T=2048.
// Layer1 (bi-LSTM): 32 persistent workgroups (16 fwd + 16 bwd), each owns a
//   16-row batch slice (recurrence is independent per batch row -> no
//   grid sync). Per step: [16x64]h @ Whh^T[64x256] via v_wmma_f32_16x16x32_bf16,
//   IN=4 input projection fused as VALU FMAs. h stored bf16 to out1 (d_ws,
//   128 MB -> L2-resident at 192 MB).
// Layer2: 16 persistent workgroups; per step K=192 (128 input + 64 hidden)
//   -> 6 WMMAs per 16-col gate tile; input A-frags read straight from
//   L2-resident bf16 out1.
// FC: trivial 256-thread kernel.
// ---------------------------------------------------------------------------

#define HH   64
#define TT   2048
#define BB   256
#define G4   256      // 4*H

typedef __attribute__((ext_vector_type(16))) __bf16 v16bf;
typedef __attribute__((ext_vector_type(8)))  float  v8f;

union FragB16 {
    v16bf          v;
    unsigned short s[16];
    uint4          q[2];
};

__device__ __forceinline__ unsigned short f2bf(float f) {
    unsigned int u = __float_as_uint(f);
    u += 0x7FFFu + ((u >> 16) & 1u);      // round-to-nearest-even
    return (unsigned short)(u >> 16);
}

__device__ __forceinline__ float fast_sigmoid(float v) {
    return 1.0f / (1.0f + __expf(-v));
}
__device__ __forceinline__ float fast_tanh(float v) {
    float e = __expf(2.0f * v);
    return (e - 1.0f) / (e + 1.0f);
}

// D = A*B + C, bf16 inputs, f32 accumulate.
__device__ __forceinline__ v8f wmma_bf16(v16bf a, v16bf b, v8f c) {
    return __builtin_amdgcn_wmma_f32_16x16x32_bf16(
        /*neg_a=*/false, a, /*neg_b=*/false, b,
        /*c_mod=*/(short)0, c, /*reuse_a=*/false, /*reuse_b=*/false);
}

// B-matrix fragment for h @ W^T : B[k][n] = W[n][k], W row-major f32 [.,ld].
// Lane L holds W row n0+(L&15); elems 0..7 = K kk..kk+7, elems 8..15 =
// K kk+16..kk+23, kk = k0 + (L>=16 ? 8 : 0).   (ISA 7.12.2, 16-bit B layout)
__device__ __forceinline__ v16bf load_bfragW(const float* __restrict__ W,
                                             int ld, int n0, int k0) {
    const int lane = threadIdx.x & 31;
    const float* q = W + (size_t)(n0 + (lane & 15)) * ld + k0 + ((lane & 16) ? 8 : 0);
    FragB16 f;
#pragma unroll
    for (int e = 0; e < 16; ++e) f.s[e] = f2bf(q[e < 8 ? e : e + 8]);
    return f.v;
}

// A-matrix fragment from row-major bf16 (LDS h buffer). Lane L = row (L&15);
// elems 0..7 = K kk..kk+7, elems 8..15 = K kk+16..kk+23, kk = k0 + (L>=16?8:0).
__device__ __forceinline__ v16bf load_afrag_lds(const unsigned short* base,
                                                int row_stride, int k0) {
    const int lane = threadIdx.x & 31;
    const unsigned short* p = base + (lane & 15) * row_stride + k0 + ((lane & 16) ? 8 : 0);
    FragB16 f;
    f.q[0] = *(const uint4*)(p);
    f.q[1] = *(const uint4*)(p + 16);
    return f.v;
}

// A-matrix fragment straight from global bf16 out1 [B][T][128].
__device__ __forceinline__ v16bf load_afrag_out1(const unsigned short* __restrict__ out1,
                                                 int b0, int t, int k0) {
    const int lane = threadIdx.x & 31;
    const unsigned short* p = out1 +
        ((size_t)(b0 + (lane & 15)) * TT + t) * 128 + k0 + ((lane & 16) ? 8 : 0);
    FragB16 f;
    f.q[0] = *(const uint4*)(p);
    f.q[1] = *(const uint4*)(p + 16);
    return f.v;
}

// ---------------------------------------------------------------------------
// Layer 1: bidirectional LSTM. grid = 32 blocks (0..15 fwd, 16..31 bwd),
// block = 256 threads (8 waves, 2 gate tiles of 16 cols each per wave).
// ---------------------------------------------------------------------------
__global__ __launch_bounds__(256) void lstm_layer1(
    const float* __restrict__ x,       // [B, T, 4]
    const float* __restrict__ Wih_f, const float* __restrict__ Whh_f, const float* __restrict__ b_f,
    const float* __restrict__ Wih_b, const float* __restrict__ Whh_b, const float* __restrict__ b_b,
    unsigned short* __restrict__ out1) // [B, T, 128] bf16 (fwd cols 0..63, bwd 64..127)
{
    __shared__ unsigned short lds_h[16][72];   // h state, bf16, padded rows
    __shared__ float          lds_g[16][260];  // raw gate pre-activations

    const int tid  = threadIdx.x;
    const int wave = tid >> 5;
    const int lane = tid & 31;
    const int dir  = (int)(blockIdx.x >> 4);
    const int b0   = ((int)blockIdx.x & 15) * 16;

    const float* Wih  = dir ? Wih_b : Wih_f;
    const float* Whh  = dir ? Whh_b : Whh_f;
    const float* bias = dir ? b_b   : b_f;

    const int n0a = wave * 32;          // two 16-col gate tiles per wave
    const int n0b = n0a + 16;

    // Persistent B fragments of Whh (K=64 -> 2 chunks per tile).
    v16bf Ba0 = load_bfragW(Whh, HH, n0a, 0);
    v16bf Ba1 = load_bfragW(Whh, HH, n0a, 32);
    v16bf Bb0 = load_bfragW(Whh, HH, n0b, 0);
    v16bf Bb1 = load_bfragW(Whh, HH, n0b, 32);

    // Per-lane fused input weights + bias for this lane's two gate columns.
    const int ca = n0a + (lane & 15);
    const int cb = n0b + (lane & 15);
    const float4 wa = *(const float4*)(Wih + (size_t)ca * 4);
    const float4 wb = *(const float4*)(Wih + (size_t)cb * 4);
    const float biasa = bias[ca];
    const float biasb = bias[cb];

    // h0 = 0
    for (int i = tid; i < 16 * 72; i += 256) ((unsigned short*)lds_h)[i] = 0;

    // Stage-2 mapping: thread -> batch row, 4 consecutive hidden cols; c in regs.
    float creg[4] = {0.f, 0.f, 0.f, 0.f};
    const int s2_b  = tid >> 4;
    const int s2_j0 = (tid & 15) << 2;

    const int mbase = (lane & 16) ? 8 : 0;

    for (int t = 0; t < TT; ++t) {
        const int tt = dir ? (TT - 1 - t) : t;
        __syncthreads();   // lds_h from previous step ready; lds_g reusable

        // ---- Stage 1: gates = h @ Whh^T + x @ Wih^T + b  (raw, into LDS) ----
        v16bf a0 = load_afrag_lds(&lds_h[0][0], 72, 0);
        v16bf a1 = load_afrag_lds(&lds_h[0][0], 72, 32);
        v8f acc0 = {};
        v8f acc1 = {};
        acc0 = wmma_bf16(a0, Ba0, acc0);
        acc0 = wmma_bf16(a1, Ba1, acc0);
        acc1 = wmma_bf16(a0, Bb0, acc1);
        acc1 = wmma_bf16(a1, Bb1, acc1);

#pragma unroll
        for (int r = 0; r < 8; ++r) {
            const int m = mbase + r;
            const float4 xv = *(const float4*)(x + ((size_t)(b0 + m) * TT + tt) * 4);
            const float prea = biasa + xv.x * wa.x + xv.y * wa.y + xv.z * wa.z + xv.w * wa.w;
            const float preb = biasb + xv.x * wb.x + xv.y * wb.y + xv.z * wb.z + xv.w * wb.w;
            lds_g[m][ca] = acc0[r] + prea;
            lds_g[m][cb] = acc1[r] + preb;
        }

        // Prefetch next timestep's x slab while gates settle.
        if (t + 1 < TT && tid < 16) {
            const int tn = dir ? (TT - 2 - t) : (t + 1);
            __builtin_prefetch(x + ((size_t)(b0 + tid) * TT + tn) * 4, 0, 3);
        }
        __syncthreads();

        // ---- Stage 2: activations, c/h update, h -> LDS + global out1 ----
        unsigned short hb[4];
#pragma unroll
        for (int k = 0; k < 4; ++k) {
            const int j = s2_j0 + k;
            const float iv = lds_g[s2_b][j];
            const float fv = lds_g[s2_b][64 + j];
            const float gv = lds_g[s2_b][128 + j];
            const float ov = lds_g[s2_b][192 + j];
            const float c  = fast_sigmoid(fv) * creg[k] + fast_sigmoid(iv) * fast_tanh(gv);
            creg[k] = c;
            const float h = fast_sigmoid(ov) * fast_tanh(c);
            hb[k] = f2bf(h);
            lds_h[s2_b][j] = hb[k];
        }
        const size_t o = ((size_t)(b0 + s2_b) * TT + tt) * 128 + dir * 64 + s2_j0;
        *(uint2*)(out1 + o) = make_uint2((unsigned)hb[0] | ((unsigned)hb[1] << 16),
                                         (unsigned)hb[2] | ((unsigned)hb[3] << 16));
    }
}

// ---------------------------------------------------------------------------
// Layer 2: unidirectional LSTM over out1 [B,T,128]. grid = 16 blocks,
// block = 512 threads (16 waves, one 16-col gate tile each). Input GEMM
// fused into the WMMA chain: K = 128 (input) + 64 (hidden) = 6 WMMAs/tile.
// ---------------------------------------------------------------------------
__global__ __launch_bounds__(512) void lstm_layer2(
    const unsigned short* __restrict__ out1, // [B,T,128] bf16
    const float* __restrict__ Wih2,          // [256][128]
    const float* __restrict__ Whh2,          // [256][64]
    const float* __restrict__ b2,            // [256]
    float* __restrict__ h2)                  // [B, 64] final hidden
{
    __shared__ unsigned short lds_h[16][72];
    __shared__ float          lds_g[16][260];

    const int tid  = threadIdx.x;
    const int wave = tid >> 5;
    const int lane = tid & 31;
    const int b0   = (int)blockIdx.x * 16;
    const int n0   = wave * 16;

    // Persistent B fragments: 4 chunks of Wih2 (K=128) + 2 of Whh2 (K=64).
    v16bf Bi0 = load_bfragW(Wih2, 128, n0, 0);
    v16bf Bi1 = load_bfragW(Wih2, 128, n0, 32);
    v16bf Bi2 = load_bfragW(Wih2, 128, n0, 64);
    v16bf Bi3 = load_bfragW(Wih2, 128, n0, 96);
    v16bf Bh0 = load_bfragW(Whh2, 64, n0, 0);
    v16bf Bh1 = load_bfragW(Whh2, 64, n0, 32);

    const int   cn    = n0 + (lane & 15);
    const float biasn = b2[cn];

    for (int i = tid; i < 16 * 72; i += 512) ((unsigned short*)lds_h)[i] = 0;

    float creg[2] = {0.f, 0.f};
    const int s2_b  = tid >> 5;
    const int s2_j0 = (tid & 31) << 1;
    const int mbase = (lane & 16) ? 8 : 0;

    for (int t = 0; t < TT; ++t) {
        __syncthreads();

        // ---- Stage 1: gates = in_t @ Wih2^T + h @ Whh2^T + b2 ----
        v8f acc = {};
        acc = wmma_bf16(load_afrag_out1(out1, b0, t, 0),  Bi0, acc);
        acc = wmma_bf16(load_afrag_out1(out1, b0, t, 32), Bi1, acc);
        acc = wmma_bf16(load_afrag_out1(out1, b0, t, 64), Bi2, acc);
        acc = wmma_bf16(load_afrag_out1(out1, b0, t, 96), Bi3, acc);
        acc = wmma_bf16(load_afrag_lds(&lds_h[0][0], 72, 0),  Bh0, acc);
        acc = wmma_bf16(load_afrag_lds(&lds_h[0][0], 72, 32), Bh1, acc);

#pragma unroll
        for (int r = 0; r < 8; ++r)
            lds_g[mbase + r][cn] = acc[r] + biasn;
        __syncthreads();

        // ---- Stage 2: activations + state update ----
#pragma unroll
        for (int k = 0; k < 2; ++k) {
            const int j = s2_j0 + k;
            const float iv = lds_g[s2_b][j];
            const float fv = lds_g[s2_b][64 + j];
            const float gv = lds_g[s2_b][128 + j];
            const float ov = lds_g[s2_b][192 + j];
            const float c  = fast_sigmoid(fv) * creg[k] + fast_sigmoid(iv) * fast_tanh(gv);
            creg[k] = c;
            const float h = fast_sigmoid(ov) * fast_tanh(c);
            lds_h[s2_b][j] = f2bf(h);
            if (t == TT - 1)
                h2[(size_t)(b0 + s2_b) * HH + j] = h;
        }
    }
}

// ---------------------------------------------------------------------------
// Final FC: logits[b] = h2[b,:] . Wfc + bfc
// ---------------------------------------------------------------------------
__global__ __launch_bounds__(256) void fc_kernel(
    const float* __restrict__ h2, const float* __restrict__ Wfc,
    const float* __restrict__ bfc, float* __restrict__ out)
{
    const int b = threadIdx.x;
    float s = bfc[0];
#pragma unroll
    for (int j = 0; j < HH; ++j)
        s += h2[(size_t)b * HH + j] * Wfc[j];
    out[b] = s;
}

// ---------------------------------------------------------------------------
extern "C" void kernel_launch(void* const* d_in, const int* in_sizes, int n_in,
                              void* d_out, int out_size, void* d_ws, size_t ws_size,
                              hipStream_t stream) {
    (void)in_sizes; (void)n_in; (void)out_size; (void)ws_size;

    const float* x     = (const float*)d_in[0];
    const float* Wih_f = (const float*)d_in[1];
    const float* Whh_f = (const float*)d_in[2];
    const float* b_f   = (const float*)d_in[3];
    const float* Wih_b = (const float*)d_in[4];
    const float* Whh_b = (const float*)d_in[5];
    const float* b_b   = (const float*)d_in[6];
    const float* Wih2  = (const float*)d_in[7];
    const float* Whh2  = (const float*)d_in[8];
    const float* b2    = (const float*)d_in[9];
    const float* Wfc   = (const float*)d_in[10];
    const float* bfc   = (const float*)d_in[11];

    // Workspace layout: out1 bf16 [B][T][128] = 128 MB, then h2 f32 [B][64].
    unsigned short* out1 = (unsigned short*)d_ws;
    float* h2 = (float*)((char*)d_ws + (size_t)BB * TT * 128 * sizeof(unsigned short));

    lstm_layer1<<<32, 256, 0, stream>>>(x, Wih_f, Whh_f, b_f,
                                        Wih_b, Whh_b, b_b, out1);
    lstm_layer2<<<16, 512, 0, stream>>>(out1, Wih2, Whh2, b2, h2);
    fc_kernel<<<1, 256, 0, stream>>>(h2, Wfc, bfc, (float*)d_out);
}
